// oMacConv2d_73469710565715
// MI455X (gfx1250) — 8-bit matrix operand order repaired and checked by reading, compiled, not run on hardware
//
#include <hip/hip_runtime.h>


typedef int v8i  __attribute__((ext_vector_type(8)));
typedef int v4ig __attribute__((vector_size(16)));

#define CIN    64
#define COUT   128
#define HH     112
#define WW     112
#define NBATCH 32
#define ROWS   4
#define ACOLS  18
#define ASTRID 80

#if defined(__AMDGCN__) && \
    __has_builtin(__builtin_amdgcn_global_load_async_to_lds_b128) && \
    __has_builtin(__builtin_amdgcn_s_wait_asynccnt)
#define HAVE_ASYNC_LDS 1
typedef __attribute__((address_space(1))) v4ig* as1_v4i;
typedef __attribute__((address_space(3))) v4ig* as3_v4i;
#endif

__global__ __launch_bounds__(256) void quant_w_kernel(const float* __restrict__ w,
                                                      signed char* __restrict__ w8) {
    int idx = blockIdx.x * 256 + threadIdx.x;
    if (idx >= COUT * 9 * CIN) return;
    int cin  = idx & 63;
    int tap  = (idx >> 6) % 9;
    int cout = idx / (9 * CIN);
    float v = w[(size_t)(cout * CIN + cin) * 9 + tap];
    v = rintf(v);
    v = fminf(fmaxf(v, -8.f), 7.f);
    w8[idx] = (signed char)(int)v;
}

__global__ __launch_bounds__(256) void quant_x_kernel(const float* __restrict__ x,
                                                      unsigned char* __restrict__ x8) {
    __shared__ __align__(16) unsigned char row8[WW * CIN];
    const int h = blockIdx.x, n = blockIdx.y, t = threadIdx.x;
    const float* src = x + ((size_t)(n * CIN) * HH + h) * WW;
    for (int i = t; i < CIN * WW; i += 256) {
        int cin = i / WW, w = i % WW;
        float v = __builtin_nontemporal_load(&src[(size_t)cin * HH * WW + w]);
        v = rintf(v);
        v = fminf(fmaxf(v, 0.f), 15.f);
        row8[w * CIN + cin] = (unsigned char)(int)v;
    }
    __syncthreads();
    uint4* dst = (uint4*)(x8 + (size_t)(n * HH + h) * (WW * CIN));
    const uint4* s = (const uint4*)row8;
    for (int i = t; i < (WW * CIN) / 16; i += 256) dst[i] = s[i];
}

__global__ __launch_bounds__(256) void conv_iu8_kernel(const unsigned char* __restrict__ x8,
                                                       const signed char*  __restrict__ w8,
                                                       float* __restrict__ out) {
    __shared__ __align__(16) unsigned char alds[(ROWS + 2) * ACOLS * ASTRID];
    const int c0 = blockIdx.x * 16;
    const int h0 = blockIdx.y * ROWS;
    const int n  = blockIdx.z;
    const int t  = threadIdx.x;

#ifdef HAVE_ASYNC_LDS
    {
        const uint4 z = make_uint4(0u, 0u, 0u, 0u);
        for (int i = t; i < ((ROWS + 2) * ACOLS * ASTRID) / 16; i += 256)
            *(uint4*)(alds + i * 16) = z;
    }
    __syncthreads();
    for (int ch = t; ch < (ROWS + 2) * ACOLS * 4; ch += 256) {
        int r    = ch / (ACOLS * 4);
        int rem  = ch % (ACOLS * 4);
        int col  = rem >> 2;
        int q    = rem & 3;
        int grow = h0 + r - 1;
        int gcol = c0 + col - 1;
        if ((unsigned)grow < HH && (unsigned)gcol < WW) {
            const unsigned char* gp =
                x8 + ((size_t)(n * HH + grow) * WW + gcol) * CIN + q * 16;
            unsigned char* lp = alds + (r * ACOLS + col) * ASTRID + q * 16;
            __builtin_amdgcn_global_load_async_to_lds_b128(
                (as1_v4i)gp, (as3_v4i)lp, 0, 0);
        }
    }
    __builtin_amdgcn_s_wait_asynccnt(0);
    __syncthreads();
#else
    for (int ch = t; ch < (ROWS + 2) * ACOLS * 4; ch += 256) {
        int r    = ch / (ACOLS * 4);
        int rem  = ch % (ACOLS * 4);
        int col  = rem >> 2;
        int q    = rem & 3;
        int grow = h0 + r - 1;
        int gcol = c0 + col - 1;
        uint4 v = make_uint4(0u, 0u, 0u, 0u);
        if ((unsigned)grow < HH && (unsigned)gcol < WW)
            v = *(const uint4*)(x8 + ((size_t)(n * HH + grow) * WW + gcol) * CIN + q * 16);
        *(uint4*)(alds + (r * ACOLS + col) * ASTRID + q * 16) = v;
    }
    __syncthreads();
#endif

    const int wv   = t >> 5;
    const int lane = t & 31;
    const int lo   = lane & 15;
    const int hi   = (lane >> 4) & 1;

    union AF { v8i v; unsigned long long q[4]; };
    AF A[9];
    const int cout = wv * 16 + lo;
#pragma unroll
    for (int tap = 0; tap < 9; ++tap) {
        const signed char* wrow = w8 + ((size_t)cout * 9 + tap) * CIN;
        A[tap].q[0] = *(const unsigned long long*)(wrow +      (hi ? 16 : 0));
        A[tap].q[1] = *(const unsigned long long*)(wrow + 8  + (hi ? 16 : 0));
        A[tap].q[2] = *(const unsigned long long*)(wrow + 32 + (hi ? 16 : 0));
        A[tap].q[3] = *(const unsigned long long*)(wrow + 40 + (hi ? 16 : 0));
    }

#pragma unroll
    for (int r = 0; r < ROWS; ++r) {
        v8i acc = {0, 0, 0, 0, 0, 0, 0, 0};
#pragma unroll
        for (int tap = 0; tap < 9; ++tap) {
            const int kh = tap / 3, kw = tap % 3;
            union BF { v8i v; uint4 x[2]; } B;
            const unsigned char* bb = alds + ((r + kh) * ACOLS + (lo + kw)) * ASTRID;
            B.x[0] = *(const uint4*)(bb +      (hi ? 16 : 0));
            B.x[1] = *(const uint4*)(bb + 32 + (hi ? 16 : 0));
            acc = __builtin_amdgcn_wmma_i32_16x16x64_iu8(true, A[tap].v,
                                                         false, B.v,
                                                         acc, false, false);
        }
        const int hOut = h0 + r;
#pragma unroll
        for (int v = 0; v < 8; ++v) {
            const int co = wv * 16 + (hi ? 8 : 0) + v;
            __builtin_nontemporal_store(
                (float)acc[v],
                &out[(((size_t)(n * COUT + co)) * HH + hOut) * WW + c0 + lo]);
        }
    }
}

extern "C" void kernel_launch(void* const* d_in, const int* in_sizes, int n_in,
                              void* d_out, int out_size, void* d_ws, size_t ws_size,
                              hipStream_t stream) {
    const float* x = (const float*)d_in[0];
    const float* w = (const float*)d_in[1];
    float* out = (float*)d_out;

    signed char*   w8 = (signed char*)d_ws;
    unsigned char* x8 = (unsigned char*)d_ws + (size_t)COUT * 9 * CIN;

    quant_w_kernel<<<(COUT * 9 * CIN + 255) / 256, 256, 0, stream>>>(w, w8);
    quant_x_kernel<<<dim3(HH, NBATCH), 256, 0, stream>>>(x, x8);
    conv_iu8_kernel<<<dim3(WW / 16, HH / ROWS, NBATCH), 256, 0, stream>>>(x8, w8, out);
}
